// Encoder_ATTENTION_53249004536172
// MI455X (gfx1250) — compile-verified, matrix operand order checked by reading, not run on hardware
//
#include <hip/hip_runtime.h>
#include <hip/hip_bf16.h>
#include <stdint.h>

// ---------------------------------------------------------------------------
// Problem constants (from reference)
// ---------------------------------------------------------------------------
#define BATCH   4096
#define NB      64
#define DIM     256
#define TWOD    512
#define CNT_E   1000          // padding relation id
#define N_WR    1001
#define N_ZQ    1000

typedef __attribute__((ext_vector_type(16))) __bf16 v16bf;
typedef __attribute__((ext_vector_type(8)))  float  v8f;

union Frag16 {                 // 8 VGPRs of packed bf16 (16 values) per lane
    v16bf     v;
    uint32_t  u[8];
    uint4     q[2];
};

// bf16 round-to-nearest-even pack helpers
__device__ __forceinline__ uint32_t f2bf(float f) {
    uint32_t u = __float_as_uint(f);
    return (u + 0x7FFFu + ((u >> 16) & 1u)) >> 16;
}
__device__ __forceinline__ uint32_t pack_bf16(float lo, float hi) {
    return f2bf(lo) | (f2bf(hi) << 16);
}

// ---------------------------------------------------------------------------
// Prep kernel 1: zq_lin[q][o] = attn_W_b[o] + sum_k zq_weight[q][k]*W[o][k]
// (first half of W's input dim multiplies z_q).  1000x512 f32 -> workspace.
// ---------------------------------------------------------------------------
__global__ void __launch_bounds__(256) zq_prep_kernel(
        const float* __restrict__ zq_w, const float* __restrict__ W,
        const float* __restrict__ bias, float* __restrict__ zq_lin) {
    __shared__ float s_z[DIM];
    const int q = blockIdx.x;
    const int t = threadIdx.x;
    s_z[t] = zq_w[(size_t)q * DIM + t];
    __syncthreads();
    #pragma unroll
    for (int half = 0; half < 2; ++half) {
        const int o = half * 256 + t;
        const float4* wr = reinterpret_cast<const float4*>(W + (size_t)o * TWOD);
        float acc = bias[o];
        #pragma unroll 4
        for (int k4 = 0; k4 < DIM / 4; ++k4) {
            float4 w4 = wr[k4];
            acc += w4.x * s_z[k4 * 4 + 0] + w4.y * s_z[k4 * 4 + 1]
                 + w4.z * s_z[k4 * 4 + 2] + w4.w * s_z[k4 * 4 + 3];
        }
        zq_lin[(size_t)q * TWOD + o] = acc;
    }
}

// ---------------------------------------------------------------------------
// Prep kernel 2: swizzle W2 = attn_W_w[:, 256:512] into per-lane bf16 B-matrix
// fragments for v_wmma_f32_16x16x32_bf16.
// Layout: Bbf[((ntile*8 + kstep)*32 + lane)*8 + vgpr] (dword = 2 bf16).
// 16-bit operand layout (ISA 7.12.2): lane<16 -> K=kbase+0..7 & +16..23
// with kbase=0, lane>=16 -> kbase=8.  B is stored N-major mirroring A.
// ---------------------------------------------------------------------------
__global__ void __launch_bounds__(256) b_prep_kernel(
        const float* __restrict__ W, uint32_t* __restrict__ Bbf) {
    const int idx  = blockIdx.x * 256 + threadIdx.x;   // 0 .. 65535
    const int v    = idx & 7;
    const int lane = (idx >> 3) & 31;
    const int ks   = (idx >> 8) & 7;
    const int nt   = idx >> 11;
    const int n     = nt * 16 + (lane & 15);
    const int kbase = (lane < 16) ? 0 : 8;
    const int k     = ks * 32 + kbase + ((v & 3) * 2) + ((v >= 4) ? 16 : 0);
    const float* row = W + (size_t)n * TWOD + DIM;     // second half of W row
    Bbf[idx] = pack_bf16(row[k], row[k + 1]);
}

// ---------------------------------------------------------------------------
// Main fused kernel: one block per batch element, 256 threads = 8 waves.
// ---------------------------------------------------------------------------
__global__ void __launch_bounds__(256) encoder_attn_kernel(
        const int*   __restrict__ rid,      // [B, NB]
        const float* __restrict__ e_emb,    // [B, NB, DIM]
        const float* __restrict__ rw,       // [B, NB]
        const int*   __restrict__ qrid,     // [B]
        const float* __restrict__ w_r,      // [N_WR, DIM]
        const float* __restrict__ ua_w,     // [512]
        const float* __restrict__ ua_b,     // [1]
        const uint32_t* __restrict__ Bbf,   // swizzled W2 bf16 fragments
        const float* __restrict__ zq_lin,   // [N_ZQ, 512]
        float* __restrict__ out)            // [B, DIM]
{
    __shared__ uint32_t s_etr[NB * (DIM / 2)];  // 64 x 256 bf16 (packed), 32 KB
    __shared__ float    s_zq[TWOD];             // zq_lin row, 2 KB
    __shared__ float    s_alpha[NB];
    __shared__ float    s_attn[NB];

    const int b    = blockIdx.x;
    const int tid  = threadIdx.x;
    const int lane = tid & 31;
    const int wave = tid >> 5;

    // --- load zq_lin row, init alpha with u_a bias -------------------------
    const int q = qrid[b];
    for (int i = tid; i < TWOD; i += 256) s_zq[i] = zq_lin[(size_t)q * TWOD + i];
    if (tid < NB) s_alpha[tid] = ua_b[0];

    // --- Phase 1: TransH projection, write e_tr as packed bf16 to LDS ------
    // wave handles neighbors wave*8 .. wave*8+7; lane covers k = lane*8..+7.
    #pragma unroll
    for (int j = 0; j < 8; ++j) {
        const int n = wave * 8 + j;
        const int r = rid[b * NB + n];
        const float4* wr4 = reinterpret_cast<const float4*>(
                                w_r + (size_t)r * DIM + lane * 8);
        const float4* ee4 = reinterpret_cast<const float4*>(
                                e_emb + ((size_t)b * NB + n) * DIM + lane * 8);
        float4 w0 = wr4[0], w1 = wr4[1];
        float4 e0 = ee4[0], e1 = ee4[1];
        float nrm2 = w0.x*w0.x + w0.y*w0.y + w0.z*w0.z + w0.w*w0.w
                   + w1.x*w1.x + w1.y*w1.y + w1.z*w1.z + w1.w*w1.w;
        float dot  = w0.x*e0.x + w0.y*e0.y + w0.z*e0.z + w0.w*e0.w
                   + w1.x*e1.x + w1.y*e1.y + w1.z*e1.z + w1.w*e1.w;
        #pragma unroll
        for (int off = 16; off > 0; off >>= 1) {
            nrm2 += __shfl_xor(nrm2, off, 32);
            dot  += __shfl_xor(dot,  off, 32);
        }
        float nrm  = fmaxf(sqrtf(nrm2), 1e-12f);
        float coef = dot / (nrm * nrm);                 // (e . w/|w|) / |w|
        float mask = (r < CNT_E) ? 1.0f : 0.0f;
        float t0 = (e0.x - coef * w0.x) * mask, t1 = (e0.y - coef * w0.y) * mask;
        float t2 = (e0.z - coef * w0.z) * mask, t3 = (e0.w - coef * w0.w) * mask;
        float t4 = (e1.x - coef * w1.x) * mask, t5 = (e1.y - coef * w1.y) * mask;
        float t6 = (e1.z - coef * w1.z) * mask, t7 = (e1.w - coef * w1.w) * mask;
        uint32_t* dst = &s_etr[n * (DIM / 2) + lane * 4];
        dst[0] = pack_bf16(t0, t1);
        dst[1] = pack_bf16(t2, t3);
        dst[2] = pack_bf16(t4, t5);
        dst[3] = pack_bf16(t6, t7);
    }
    __syncthreads();

    // --- Phase 2: e_tr[64x256] @ W2^T[256x512] via bf16 WMMA ---------------
    // M tiles: 4 (neighbors), N tiles: 32 (outputs), K steps: 8 (x32).
    // wave owns N tiles [wave*4, wave*4+4).
    const int nt0    = wave * 4;
    const int mrow   = lane & 15;
    const int khalf  = (lane < 16) ? 0 : 4;   // dword offset for kbase 0/8

    v8f acc[4][4];
    #pragma unroll
    for (int mt = 0; mt < 4; ++mt)
        #pragma unroll
        for (int ntl = 0; ntl < 4; ++ntl)
            acc[mt][ntl] = (v8f){0.f, 0.f, 0.f, 0.f, 0.f, 0.f, 0.f, 0.f};

    for (int ks = 0; ks < 8; ++ks) {
        Frag16 afr[4];
        #pragma unroll
        for (int mt = 0; mt < 4; ++mt) {
            const uint32_t* arow =
                &s_etr[(mt * 16 + mrow) * (DIM / 2) + ks * 16 + khalf];
            afr[mt].q[0] = *reinterpret_cast<const uint4*>(arow);      // K lo 8
            afr[mt].q[1] = *reinterpret_cast<const uint4*>(arow + 8);  // K hi 8
        }
        #pragma unroll
        for (int ntl = 0; ntl < 4; ++ntl) {
            const int nt = nt0 + ntl;
            const uint4* bp = reinterpret_cast<const uint4*>(
                Bbf + ((((size_t)nt * 8 + ks) * 32 + lane) << 3));
            Frag16 bfr;
            bfr.q[0] = bp[0];
            bfr.q[1] = bp[1];
            #pragma unroll
            for (int mt = 0; mt < 4; ++mt) {
                acc[mt][ntl] = __builtin_amdgcn_wmma_f32_16x16x32_bf16(
                    /*neg_a=*/false, afr[mt].v,
                    /*neg_b=*/false, bfr.v,
                    /*c_mod=*/(short)0, acc[mt][ntl],
                    /*reuse_a=*/false, /*reuse_b=*/false);
            }
        }
    }

    // --- Epilogue: alpha[m] += sum_o ua[o]*tanh(D[m][o] + zq_lin[o]) -------
    // C/D layout: lane<16 -> M=r, lane>=16 -> M=r+8, N = lane&15 (per tile).
    {
        const int ncol = lane & 15;
        float uav[4], zqv[4];
        #pragma unroll
        for (int ntl = 0; ntl < 4; ++ntl) {
            const int o = (nt0 + ntl) * 16 + ncol;
            uav[ntl] = ua_w[o];
            zqv[ntl] = s_zq[o];
        }
        #pragma unroll
        for (int mt = 0; mt < 4; ++mt) {
            #pragma unroll
            for (int r = 0; r < 8; ++r) {
                float part = 0.f;
                #pragma unroll
                for (int ntl = 0; ntl < 4; ++ntl)
                    part += uav[ntl] * tanhf(acc[mt][ntl][r] + zqv[ntl]);
                // reduce over the 16 lanes sharing this M row
                part += __shfl_xor(part, 8, 32);
                part += __shfl_xor(part, 4, 32);
                part += __shfl_xor(part, 2, 32);
                part += __shfl_xor(part, 1, 32);
                if (ncol == 0) {
                    const int m = mt * 16 + r + ((lane < 16) ? 0 : 8);
                    atomicAdd(&s_alpha[m], part);
                }
            }
        }
    }
    __syncthreads();

    // --- Phase 3: masked softmax over NB + rw (wave 0) ---------------------
    if (wave == 0) {
        float a0 = s_alpha[lane];
        float a1 = s_alpha[lane + 32];
        const int r0 = rid[b * NB + lane];
        const int r1 = rid[b * NB + lane + 32];
        if (r0 == CNT_E) a0 -= 1e19f;
        if (r1 == CNT_E) a1 -= 1e19f;
        float mx = fmaxf(a0, a1);
        #pragma unroll
        for (int off = 16; off > 0; off >>= 1)
            mx = fmaxf(mx, __shfl_xor(mx, off, 32));
        float e0 = __expf(a0 - mx);
        float e1 = __expf(a1 - mx);
        float sm = e0 + e1;
        #pragma unroll
        for (int off = 16; off > 0; off >>= 1)
            sm += __shfl_xor(sm, off, 32);
        const float inv = 1.0f / sm;
        s_attn[lane]      = e0 * inv + rw[b * NB + lane];
        s_attn[lane + 32] = e1 * inv + rw[b * NB + lane + 32];
    }
    __syncthreads();

    // --- Phase 4: out[b][d] = sum_n attn[n] * e_tr[n][d] -------------------
    {
        const int d = tid;            // 256 threads cover DIM
        float accd = 0.f;
        #pragma unroll 8
        for (int n = 0; n < NB; ++n) {
            const uint32_t u = s_etr[n * (DIM / 2) + (d >> 1)];
            const float v = __uint_as_float((d & 1) ? (u & 0xFFFF0000u)
                                                    : (u << 16));
            accd += s_attn[n] * v;
        }
        out[(size_t)b * DIM + d] = accd;
    }
}

// ---------------------------------------------------------------------------
// Launch
// ---------------------------------------------------------------------------
extern "C" void kernel_launch(void* const* d_in, const int* in_sizes, int n_in,
                              void* d_out, int out_size, void* d_ws, size_t ws_size,
                              hipStream_t stream) {
    const int*   rid     = (const int*)  d_in[0];
    const float* e_emb   = (const float*)d_in[1];
    const float* rw      = (const float*)d_in[2];
    const int*   qrid    = (const int*)  d_in[3];
    const float* w_r     = (const float*)d_in[4];
    const float* zq_w    = (const float*)d_in[5];
    const float* attn_W  = (const float*)d_in[6];
    const float* attn_b  = (const float*)d_in[7];
    const float* ua_w    = (const float*)d_in[8];
    const float* ua_b    = (const float*)d_in[9];
    float*       out     = (float*)d_out;

    // workspace partition: zq_lin (1000*512 f32 = 2 MB), then Bbf (256 KB)
    float*    zq_lin = (float*)d_ws;
    uint32_t* Bbf    = (uint32_t*)((char*)d_ws + (size_t)N_ZQ * TWOD * sizeof(float));

    zq_prep_kernel<<<N_ZQ, 256, 0, stream>>>(zq_w, attn_W, attn_b, zq_lin);
    b_prep_kernel<<<(32 * 8 * 32 * 8) / 256, 256, 0, stream>>>(attn_W, Bbf);
    encoder_attn_kernel<<<BATCH, 256, 0, stream>>>(
        rid, e_emb, rw, qrid, w_r, ua_w, ua_b, Bbf, zq_lin, out);
}
